// SIA_36919538877121
// MI455X (gfx1250) — matrix operand order repaired and checked by reading, not compiled, not run on hardware
//
#include <hip/hip_runtime.h>
#include <hip/hip_bf16.h>
#include <math.h>

#define BB 8
#define LL 2048
#define DD 2048
#define RR 256
#define KC 255            // toeplitz kernel size
#define BL (BB*LL)        // 16384 rows

typedef __attribute__((ext_vector_type(16))) __bf16 v16bf;
typedef __attribute__((ext_vector_type(8)))  float  v8f;

union AFrag { __bf16 e[16]; uint4 q[2]; v16bf v; };
union BFrag { uint4 q[2]; v16bf v; };

// ---- CDNA5 async-copy helpers (ASYNCcnt-tracked global->LDS DMA path) ------
typedef __attribute__((address_space(3))) char lds_char_t;

__device__ __forceinline__ uint32_t lds_off_of(const void* p) {
  // generic -> LDS address space; ptrtoint of AS(3) pointer = LDS byte offset
  return (uint32_t)(uintptr_t)(lds_char_t*)p;
}

__device__ __forceinline__ void async_b128(uint32_t lds_byte, unsigned long long gaddr) {
  asm volatile("global_load_async_to_lds_b128 %0, %1, off"
               :: "v"(lds_byte), "v"(gaddr) : "memory");
}

// 256 threads cooperatively stage one contiguous 16KB tile into LDS.
__device__ __forceinline__ void async_stage_16k(uint32_t lds_base,
                                                const void* gbase, int tid) {
#pragma unroll
  for (int i = 0; i < 4; ++i) {
    uint32_t off = (uint32_t)tid * 16u + (uint32_t)i * 4096u;
    async_b128(lds_base + off, (unsigned long long)gbase + off);
  }
}

__device__ __forceinline__ void wait_async0() {
  asm volatile("s_wait_asynccnt 0x0" ::: "memory");
}

// Packed WMMA-B fragment layout for a K x N bf16 matrix, tiled 32(K) x 16(N).
// Flat order: [kt][nt][lane][16 bf16]. Lane = ((k>>3)&1)*16|(n&15); element = (k&7)|((k>>4)&1)*8.
__device__ __forceinline__ size_t packIdx(int ntiles, int k, int n) {
  int kt = k >> 5, nt = n >> 4;
  int kk = k & 31, nn = n & 15;
  int lane = (((kk >> 3) & 1) << 4) | nn;
  int e = (kk & 7) | ((kk >> 4) << 3);
  return ((size_t)(kt * ntiles + nt) * 32 + (size_t)lane) * 16 + (size_t)e;
}

// ---------------- Phase A: per-row LayerNorm stats + row feature -------------
__global__ void k_rowstats(const float* __restrict__ x,
                           const float* __restrict__ gamma,
                           const float* __restrict__ beta,
                           float* __restrict__ meanA,
                           float* __restrict__ rstdA,
                           float* __restrict__ featA) {
  __shared__ float red[256];
  const int row = blockIdx.x;
  const int t = threadIdx.x;
  const float* xr = x + (size_t)row * DD;
  float s1 = 0.f, s2 = 0.f, sg = 0.f, sG = 0.f, sB = 0.f;
  for (int i = t; i < DD; i += 256) {
    float xv = xr[i], gv = gamma[i], bv = beta[i];
    s1 += xv; s2 += xv * xv; sg += xv * gv; sG += gv; sB += bv;
  }
  float vals[5] = {s1, s2, sg, sG, sB};
  float outv[5];
#pragma unroll
  for (int j = 0; j < 5; ++j) {
    red[t] = vals[j];
    __syncthreads();
    for (int s = 128; s > 0; s >>= 1) {
      if (t < s) red[t] += red[t + s];
      __syncthreads();
    }
    outv[j] = red[0];
    __syncthreads();
  }
  if (t == 0) {
    float mean = outv[0] / DD;
    float var  = outv[1] / DD - mean * mean;
    float rstd = rsqrtf(var + 1e-5f);
    meanA[row] = mean;
    rstdA[row] = rstd;
    featA[row] = (outv[2] - mean * outv[3]) * rstd / DD + outv[4] / DD;
  }
}

// ---------------- Phase B: per-feature gate*scale constants ------------------
__global__ void k_scale(const float* __restrict__ featA,
                        const float* __restrict__ phi_bias,
                        const float* __restrict__ phi_w,
                        const float* __restrict__ bq,
                        const float* __restrict__ g_logit,
                        const int* __restrict__ stepp,
                        float* __restrict__ c0, float* __restrict__ c1) {
  int d = blockIdx.x * 256 + threadIdx.x;
  if (d >= DD) return;
  float det_scale = fminf((float)stepp[0] / 2000.0f, 1.0f);
  float fb = 0.f;
#pragma unroll
  for (int b = 0; b < BB; ++b) fb += featA[b * LL + d];
  fb *= (1.0f / BB);
  float sdet = 0.f, sbias = 0.f, sw = 0.f;
  const float c = 3.14159265358979323846f * (d + 0.5f) / (float)LL;
  for (int r = 0; r < RR; ++r) {
    sdet  += __cosf(c * (float)r);
    sbias += phi_bias[(size_t)d * RR + r];
    sw    += phi_w[r];
  }
  float scale = (det_scale * sdet + sbias + fb * sw) * (1.0f / RR);
  float gl = g_logit[d < (DD / 2) ? 0 : 1];
  float g = 1.0f / (1.0f + __expf(-gl));
  float v1 = g * scale;
  c1[d] = v1;
  c0[d] = v1 * bq[d];
}

// ---------------- Phase C: fold toeplitz conv into dlt_T ---------------------
__global__ void k_dltc(const float* __restrict__ dlt_T,
                       const float* __restrict__ toep,
                       float* __restrict__ dltc) {
  int s = blockIdx.x, r = threadIdx.x;
  float acc = 0.f;
  for (int k = 0; k < KC; ++k) {
    int idx = s + k - (KC / 2);
    if (idx >= 0 && idx < RR) acc += toep[k] * dlt_T[(size_t)idx * RR + r];
  }
  dltc[(size_t)s * RR + r] = acc;
}

// ---------------- Phase D: pack P into WMMA-B fragments (bf16) ---------------
__global__ void k_pack_p(const float* __restrict__ P, __bf16* __restrict__ Pp) {
  int tid = blockIdx.x * 256 + threadIdx.x;  // 0 .. DD*RR-1
  int k = tid / RR, n = tid % RR;            // P is [D][R] row-major
  Pp[packIdx(RR / 16, k, n)] = (__bf16)P[tid];
}

// ---------------- Phase E: Weff = dltc^T @ Wq, packed to WMMA-B (bf16) -------
__global__ void k_weff(const float* __restrict__ dltc,
                       const float* __restrict__ Wq,
                       __bf16* __restrict__ Wp) {
  int d = blockIdx.x * 256 + threadIdx.x;
  int r = blockIdx.y;
  float acc = 0.f;
  for (int s = 0; s < RR; ++s)
    acc += dltc[(size_t)s * RR + r] * Wq[(size_t)s * DD + d];
  Wp[packIdx(DD / 16, r, d)] = (__bf16)acc;
}

// ---------------- Phase F: GEMM1  u = LN(x) @ P  (fused LN, async B staging) -
__global__ void __launch_bounds__(256) k_gemm1(
    const float* __restrict__ x, const float* __restrict__ gamma,
    const float* __restrict__ beta, const float* __restrict__ meanA,
    const float* __restrict__ rstdA, const __bf16* __restrict__ Pp,
    __bf16* __restrict__ u) {
  __shared__ __align__(16) __bf16 sB[2][16 * 512];  // 2 x 16KB B-tile buffers
  __shared__ __align__(16) float  sG[DD];
  __shared__ __align__(16) float  sBt[DD];

  const int tid  = threadIdx.x;
  const int wave = tid >> 5;
  const int lane = tid & 31;
  const int m = lane & 15, h = lane >> 4;
  const int m0 = blockIdx.x * 128 + wave * 16;
  const int row = m0 + m;
  const float* xr = x + (size_t)row * DD;
  const float mean = meanA[row], rstd = rstdA[row];

  // one-time gamma/beta staging into LDS (16KB)
  for (int i = tid; i < DD / 4; i += 256) {
    ((float4*)sG)[i]  = ((const float4*)gamma)[i];
    ((float4*)sBt)[i] = ((const float4*)beta)[i];
  }
  // prologue: async-stage B tile for kt=0
  const char* Ppb = (const char*)Pp;
  async_stage_16k(lds_off_of(&sB[0][0]), Ppb, tid);

  v8f acc[16];
#pragma unroll
  for (int i = 0; i < 16; ++i) acc[i] = (v8f){0, 0, 0, 0, 0, 0, 0, 0};

  for (int kt = 0; kt < DD / 32; ++kt) {
    wait_async0();        // this wave's async stores into LDS have landed
    __syncthreads();      // all waves' stores landed + prev compute done
    if (kt + 1 < DD / 32)
      async_stage_16k(lds_off_of(&sB[(kt + 1) & 1][0]),
                      Ppb + (size_t)(kt + 1) * 16384, tid);

    const int kb = kt * 32 + h * 8;
    if (kt + 1 < DD / 32) __builtin_prefetch(xr + kb + 32, 0, 0);
    float4 x0 = *(const float4*)(xr + kb);
    float4 x1 = *(const float4*)(xr + kb + 4);
    float4 x2 = *(const float4*)(xr + kb + 16);
    float4 x3 = *(const float4*)(xr + kb + 20);
    float4 g0 = *(const float4*)(sG + kb);
    float4 g1 = *(const float4*)(sG + kb + 4);
    float4 g2 = *(const float4*)(sG + kb + 16);
    float4 g3 = *(const float4*)(sG + kb + 20);
    float4 b0 = *(const float4*)(sBt + kb);
    float4 b1 = *(const float4*)(sBt + kb + 4);
    float4 b2 = *(const float4*)(sBt + kb + 16);
    float4 b3 = *(const float4*)(sBt + kb + 20);
    float xv[16] = {x0.x, x0.y, x0.z, x0.w, x1.x, x1.y, x1.z, x1.w,
                    x2.x, x2.y, x2.z, x2.w, x3.x, x3.y, x3.z, x3.w};
    float gv[16] = {g0.x, g0.y, g0.z, g0.w, g1.x, g1.y, g1.z, g1.w,
                    g2.x, g2.y, g2.z, g2.w, g3.x, g3.y, g3.z, g3.w};
    float bv[16] = {b0.x, b0.y, b0.z, b0.w, b1.x, b1.y, b1.z, b1.w,
                    b2.x, b2.y, b2.z, b2.w, b3.x, b3.y, b3.z, b3.w};
    AFrag a;
#pragma unroll
    for (int e = 0; e < 16; ++e)
      a.e[e] = (__bf16)((xv[e] - mean) * rstd * gv[e] + bv[e]);

    const uint4* Bq = (const uint4*)&sB[kt & 1][0];
#pragma unroll
    for (int nt = 0; nt < 16; ++nt) {
      BFrag bfr;
      bfr.q[0] = Bq[nt * 64 + lane * 2];
      bfr.q[1] = Bq[nt * 64 + lane * 2 + 1];
      acc[nt] = __builtin_amdgcn_wmma_f32_16x16x32_bf16(
          false, a.v, false, bfr.v, (short)0, acc[nt], false, false);
    }
  }

  // store u (bf16 row-major): C layout -> row = m0 + v + 8h, col = nt*16 + m
#pragma unroll
  for (int nt = 0; nt < 16; ++nt) {
    const int col = nt * 16 + m;
#pragma unroll
    for (int v = 0; v < 8; ++v) {
      const int orow = m0 + v + 8 * h;
      u[(size_t)orow * RR + col] = (__bf16)acc[nt][v];
    }
  }
}

// ---------------- Phase G: GEMM2  out = c1 * (u @ Weff) + c0 -----------------
__global__ void __launch_bounds__(256) k_gemm2(
    const __bf16* __restrict__ u, const __bf16* __restrict__ Wp,
    const float* __restrict__ c0, const float* __restrict__ c1,
    float* __restrict__ out) {
  __shared__ __align__(16) __bf16 sB[2][16 * 512];  // 2 x 16KB B-tile buffers

  const int tid  = threadIdx.x;
  const int wave = tid >> 5;
  const int lane = tid & 31;
  const int m = lane & 15, h = lane >> 4;
  const int m0 = blockIdx.x * 128 + wave * 16;
  const int n0 = blockIdx.y * 256;
  const int row = m0 + m;
  const __bf16* ur = u + (size_t)row * RR;

  const char* Wpb = (const char*)Wp;
  const int nt0 = n0 >> 4;  // 16 fragments per k-step, contiguous in packed layout
  async_stage_16k(lds_off_of(&sB[0][0]), Wpb + (size_t)nt0 * 1024, tid);

  v8f acc[16];
#pragma unroll
  for (int i = 0; i < 16; ++i) acc[i] = (v8f){0, 0, 0, 0, 0, 0, 0, 0};

#pragma unroll
  for (int kt = 0; kt < RR / 32; ++kt) {
    wait_async0();
    __syncthreads();
    if (kt + 1 < RR / 32)
      async_stage_16k(lds_off_of(&sB[(kt + 1) & 1][0]),
                      Wpb + ((size_t)(kt + 1) * (DD / 16) + nt0) * 1024, tid);

    const int kb = kt * 32 + h * 8;
    AFrag a;
    a.q[0] = *(const uint4*)(ur + kb);
    a.q[1] = *(const uint4*)(ur + kb + 16);

    const uint4* Bq = (const uint4*)&sB[kt & 1][0];
#pragma unroll
    for (int nt = 0; nt < 16; ++nt) {
      BFrag bfr;
      bfr.q[0] = Bq[nt * 64 + lane * 2];
      bfr.q[1] = Bq[nt * 64 + lane * 2 + 1];
      acc[nt] = __builtin_amdgcn_wmma_f32_16x16x32_bf16(
          false, a.v, false, bfr.v, (short)0, acc[nt], false, false);
    }
  }

#pragma unroll
  for (int nt = 0; nt < 16; ++nt) {
    const int col = n0 + nt * 16 + m;
    const float s1 = c1[col], s0 = c0[col];
#pragma unroll
    for (int v = 0; v < 8; ++v) {
      const int orow = m0 + v + 8 * h;
      out[(size_t)orow * DD + col] = acc[nt][v] * s1 + s0;
    }
  }
}

extern "C" void kernel_launch(void* const* d_in, const int* in_sizes, int n_in,
                              void* d_out, int out_size, void* d_ws, size_t ws_size,
                              hipStream_t stream) {
  const float* x        = (const float*)d_in[0];
  const float* ln_gamma = (const float*)d_in[1];
  const float* ln_beta  = (const float*)d_in[2];
  const float* P_shared = (const float*)d_in[3];
  const float* dlt_T    = (const float*)d_in[4];
  const float* phi_w    = (const float*)d_in[5];
  const float* phi_bias = (const float*)d_in[6];
  const float* toep     = (const float*)d_in[7];
  const float* Wq       = (const float*)d_in[8];
  const float* bq       = (const float*)d_in[9];
  const float* g_logit  = (const float*)d_in[10];
  const int*   step     = (const int*)d_in[11];
  float* out = (float*)d_out;

  char* ws = (char*)d_ws;
  size_t off = 0;
  auto alloc = [&](size_t bytes) -> void* {
    off = (off + 255) & ~(size_t)255;
    void* p = ws + off;
    off += bytes;
    return p;
  };
  float*  meanA = (float*)alloc((size_t)BL * 4);
  float*  rstdA = (float*)alloc((size_t)BL * 4);
  float*  featA = (float*)alloc((size_t)BL * 4);
  float*  c0    = (float*)alloc((size_t)DD * 4);
  float*  c1    = (float*)alloc((size_t)DD * 4);
  float*  dltc  = (float*)alloc((size_t)RR * RR * 4);
  __bf16* Pp    = (__bf16*)alloc((size_t)DD * RR * 2);
  __bf16* Wp    = (__bf16*)alloc((size_t)RR * DD * 2);
  __bf16* u     = (__bf16*)alloc((size_t)BL * RR * 2);
  (void)ws_size; (void)in_sizes; (void)n_in; (void)out_size;

  k_rowstats<<<BL, 256, 0, stream>>>(x, ln_gamma, ln_beta, meanA, rstdA, featA);
  k_scale<<<DD / 256, 256, 0, stream>>>(featA, phi_bias, phi_w, bq, g_logit, step, c0, c1);
  k_dltc<<<RR, RR, 0, stream>>>(dlt_T, toep, dltc);
  k_pack_p<<<(DD * RR) / 256, 256, 0, stream>>>(P_shared, Pp);
  k_weff<<<dim3(DD / 256, RR), 256, 0, stream>>>(dltc, Wq, Wp);
  k_gemm1<<<BL / 128, 256, 0, stream>>>(x, ln_gamma, ln_beta, meanA, rstdA, Pp, u);
  k_gemm2<<<dim3(BL / 128, DD / 256), 256, 0, stream>>>(u, Wp, c0, c1, out);
}
